// Block_56521769615854
// MI455X (gfx1250) — matrix operand order repaired and checked by reading, not compiled, not run on hardware
//
#include <hip/hip_runtime.h>
#include <hip/hip_bf16.h>
#include <stdint.h>

// RWKV-4 block for MI455X (gfx1250): all GEMMs on v_wmma_f32_16x16x32_bf16,
// global->LDS staging via GLOBAL_LOAD_ASYNC_TO_LDS_B128 (ASYNCcnt) with
// double-buffered LDS, BLK_K=64 (16 WMMAs per barrier). B=8,T=2048,C=1024,F=4096.

#define BB 8
#define TT 2048
#define CC 1024
#define FF 4096
#define NROWS (BB * TT)

typedef __attribute__((ext_vector_type(16))) __bf16 v16bf;
typedef __attribute__((ext_vector_type(8)))  float  v8f;

__device__ __forceinline__ unsigned short f32_to_bf16(float f) {
  unsigned int u = __float_as_uint(f);
  u += 0x7fffu + ((u >> 16) & 1u);   // round-to-nearest-even
  return (unsigned short)(u >> 16);
}

// gfx1250 async DMA: global -> LDS, 16B per lane, tracked by ASYNCcnt.
__device__ __forceinline__ void async_b128(const unsigned short* g, unsigned int* l) {
  unsigned int loff = (unsigned int)(uintptr_t)l;           // low 32b = LDS offset
  unsigned long long ga = (unsigned long long)(uintptr_t)g;
  asm volatile("global_load_async_to_lds_b128 %0, %1, off"
               :: "v"(loff), "v"(ga) : "memory");
}
__device__ __forceinline__ void wait_async0() {
  asm volatile("s_wait_asynccnt 0x0" ::: "memory");
}

// ---------------------------------------------------------------------------
// f32 -> bf16 conversion (weights), 4 elems/thread
// ---------------------------------------------------------------------------
__global__ __launch_bounds__(256) void cvt_bf16_kernel(const float* __restrict__ in,
                                                       unsigned short* __restrict__ out,
                                                       int n4) {
  int i = blockIdx.x * 256 + threadIdx.x;
  if (i < n4) {
    float4 v = reinterpret_cast<const float4*>(in)[i];
    ushort4 o;
    o.x = f32_to_bf16(v.x); o.y = f32_to_bf16(v.y);
    o.z = f32_to_bf16(v.z); o.w = f32_to_bf16(v.w);
    reinterpret_cast<ushort4*>(out)[i] = o;
  }
}

// ---------------------------------------------------------------------------
// Row LayerNorm over C=1024. One block (256 thr) per row; float4 per thread.
// ---------------------------------------------------------------------------
__global__ __launch_bounds__(256) void ln_kernel(const float* __restrict__ x,
                                                 const float* __restrict__ w,
                                                 const float* __restrict__ b,
                                                 float* __restrict__ out) {
  __shared__ float s1[256];
  __shared__ float s2[256];
  const int tid = threadIdx.x;
  const size_t row = blockIdx.x;
  float4 xv = reinterpret_cast<const float4*>(x + row * CC)[tid];
  float s = xv.x + xv.y + xv.z + xv.w;
  float q = xv.x * xv.x + xv.y * xv.y + xv.z * xv.z + xv.w * xv.w;
  s1[tid] = s; s2[tid] = q;
  __syncthreads();
  for (int off = 128; off >= 1; off >>= 1) {
    if (tid < off) { s1[tid] += s1[tid + off]; s2[tid] += s2[tid + off]; }
    __syncthreads();
  }
  const float mu  = s1[0] * (1.0f / CC);
  const float var = s2[0] * (1.0f / CC) - mu * mu;
  const float rstd = rsqrtf(var + 1e-5f);
  float4 w4 = reinterpret_cast<const float4*>(w)[tid];
  float4 b4 = reinterpret_cast<const float4*>(b)[tid];
  float4 o;
  o.x = (xv.x - mu) * rstd * w4.x + b4.x;
  o.y = (xv.y - mu) * rstd * w4.y + b4.y;
  o.z = (xv.z - mu) * rstd * w4.z + b4.z;
  o.w = (xv.w - mu) * rstd * w4.w + b4.w;
  reinterpret_cast<float4*>(out + row * CC)[tid] = o;
}

// ---------------------------------------------------------------------------
// time-mix: x* = h*tm + h[t-1]*(1-tm), bf16 outputs
// ---------------------------------------------------------------------------
__device__ __forceinline__ ushort4 mix_pack(float4 hc, float4 hp, float4 m) {
  ushort4 o;
  o.x = f32_to_bf16(hc.x * m.x + hp.x * (1.0f - m.x));
  o.y = f32_to_bf16(hc.y * m.y + hp.y * (1.0f - m.y));
  o.z = f32_to_bf16(hc.z * m.z + hp.z * (1.0f - m.z));
  o.w = f32_to_bf16(hc.w * m.w + hp.w * (1.0f - m.w));
  return o;
}

__global__ __launch_bounds__(256) void mix_att_kernel(
    const float* __restrict__ h,
    const float* __restrict__ tmk, const float* __restrict__ tmv,
    const float* __restrict__ tmr,
    unsigned short* __restrict__ xk, unsigned short* __restrict__ xv,
    unsigned short* __restrict__ xr) {
  const size_t row = blockIdx.x;
  const int t = (int)(row % TT);
  const int c4 = threadIdx.x;
  float4 hc = reinterpret_cast<const float4*>(h + row * CC)[c4];
  float4 hp = (t == 0) ? make_float4(0.f, 0.f, 0.f, 0.f)
                       : reinterpret_cast<const float4*>(h + (row - 1) * CC)[c4];
  float4 mk = reinterpret_cast<const float4*>(tmk)[c4];
  float4 mv = reinterpret_cast<const float4*>(tmv)[c4];
  float4 mr = reinterpret_cast<const float4*>(tmr)[c4];
  reinterpret_cast<ushort4*>(xk + row * CC)[c4] = mix_pack(hc, hp, mk);
  reinterpret_cast<ushort4*>(xv + row * CC)[c4] = mix_pack(hc, hp, mv);
  reinterpret_cast<ushort4*>(xr + row * CC)[c4] = mix_pack(hc, hp, mr);
}

__global__ __launch_bounds__(256) void mix_ffn_kernel(
    const float* __restrict__ h,
    const float* __restrict__ tmk, const float* __restrict__ tmr,
    unsigned short* __restrict__ xk, unsigned short* __restrict__ xr) {
  const size_t row = blockIdx.x;
  const int t = (int)(row % TT);
  const int c4 = threadIdx.x;
  float4 hc = reinterpret_cast<const float4*>(h + row * CC)[c4];
  float4 hp = (t == 0) ? make_float4(0.f, 0.f, 0.f, 0.f)
                       : reinterpret_cast<const float4*>(h + (row - 1) * CC)[c4];
  float4 mk = reinterpret_cast<const float4*>(tmk)[c4];
  float4 mr = reinterpret_cast<const float4*>(tmr)[c4];
  reinterpret_cast<ushort4*>(xk + row * CC)[c4] = mix_pack(hc, hp, mk);
  reinterpret_cast<ushort4*>(xr + row * CC)[c4] = mix_pack(hc, hp, mr);
}

// ---------------------------------------------------------------------------
// WKV recurrence (numerically stable), fused sr*y -> bf16.
// One thread per (b, c) channel; sequential over T. 8192 lanes total.
// ---------------------------------------------------------------------------
__global__ __launch_bounds__(256) void wkv_kernel(
    const float* __restrict__ time_decay, const float* __restrict__ time_first,
    const float* __restrict__ k, const float* __restrict__ v,
    const float* __restrict__ sr, unsigned short* __restrict__ rwkv) {
  const int g = blockIdx.x * 256 + threadIdx.x;   // 0..B*C-1
  const int b = g >> 10;
  const int c = g & (CC - 1);
  const float w = -__expf(time_decay[c]);
  const float u = time_first[c];
  size_t idx = (size_t)b * TT * CC + c;
  float aa = 0.f, bb = 0.f, pp = -1e38f;
  for (int t = 0; t < TT; ++t, idx += CC) {
    const float kt = k[idx];
    const float vt = v[idx];
    float ww = u + kt;
    float qq = fmaxf(pp, ww);
    float e1 = __expf(pp - qq);
    float e2 = __expf(ww - qq);
    float y = (e1 * aa + e2 * vt) / (e1 * bb + e2);
    rwkv[idx] = f32_to_bf16(sr[idx] * y);
    float ww2 = pp + w;
    float qq2 = fmaxf(ww2, kt);
    float e1b = __expf(ww2 - qq2);
    float e2b = __expf(kt - qq2);
    aa = e1b * aa + e2b * vt;
    bb = e1b * bb + e2b;
    pp = qq2;
  }
}

// ---------------------------------------------------------------------------
// final: out += sigmoid_rec * kv (elementwise, float4)
// ---------------------------------------------------------------------------
__global__ __launch_bounds__(256) void final_kernel(float* __restrict__ out,
                                                    const float* __restrict__ rr,
                                                    const float* __restrict__ kv) {
  const size_t i = (size_t)blockIdx.x * 256 + threadIdx.x;
  float4 o = reinterpret_cast<float4*>(out)[i];
  float4 r = reinterpret_cast<const float4*>(rr)[i];
  float4 q = reinterpret_cast<const float4*>(kv)[i];
  o.x += r.x * q.x; o.y += r.y * q.y; o.z += r.z * q.z; o.w += r.w * q.w;
  reinterpret_cast<float4*>(out)[i] = o;
}

// ---------------------------------------------------------------------------
// Tiled NT GEMM, bf16 in / f32 accumulate via WMMA.
//   out[m,n] = epilogue( sum_k A[m,k]*B[n,k] )
// A: [M,K] bf16 row-major, B: [N,K] bf16 row-major (both K-contiguous).
// Block: 256 thr = 8 waves. Tile 128x128x64; wave owns 64x32 -> 4x2x2 = 16 WMMA
// per k-step. Global->LDS via async-DMA (ASYNCcnt), double-buffered: one
// barrier per 64-deep k-step. LDS rows = 36 dwords (144B, 16B-aligned) so all
// fragment reads are ds_read_b128.
// EPI: 0 = f32 store, 1 = sigmoid f32, 2 = relu^2 -> bf16, 3 = res + acc f32
// ---------------------------------------------------------------------------
#define BLK_M 128
#define BLK_N 128
#define BLK_K 64
#define LDSW 36   // dwords per LDS row (32 data + 4 pad); 144B stride

union FragBF { unsigned int u[8]; v16bf v; };

template <int EPI>
__global__ __launch_bounds__(256) void gemm_nt_bf16(
    const unsigned short* __restrict__ A, const unsigned short* __restrict__ Bm,
    float* __restrict__ outF, unsigned short* __restrict__ outB,
    const float* __restrict__ res, int M, int N, int K) {
  __shared__ unsigned int lA[2][BLK_M * LDSW];
  __shared__ unsigned int lB[2][BLK_N * LDSW];

  const int tid  = threadIdx.x;
  const int lane = tid & 31;
  const int wave = tid >> 5;
  const int wm = wave >> 2;          // 0..1 -> 64-row slab
  const int wn = wave & 3;           // 0..3 -> 32-col slab
  const int bm = blockIdx.y * BLK_M;
  const int bn = blockIdx.x * BLK_N;
  const int laneHalf = lane >> 4;    // 0/1
  const int lanelo   = lane & 15;

  v8f acc[4][2];
#pragma unroll
  for (int i = 0; i < 4; ++i)
#pragma unroll
    for (int j = 0; j < 2; ++j)
#pragma unroll
      for (int e = 0; e < 8; ++e) acc[i][j][e] = 0.0f;

  // staging: tile = 128 rows x 128B = 1024 x 16B chunks per matrix,
  // 4 chunks per thread per matrix. row = c>>3, 16B-chunk-in-row = c&7.
  int srow[4], sq[4], sd[4];
#pragma unroll
  for (int i = 0; i < 4; ++i) {
    const int c = tid + i * 256;
    srow[i] = c >> 3;
    sq[i]   = c & 7;
    sd[i]   = srow[i] * LDSW + sq[i] * 4;
  }

  auto stage = [&](int buf, int k0) {
#pragma unroll
    for (int i = 0; i < 4; ++i)
      async_b128(A + (size_t)(bm + srow[i]) * K + k0 + sq[i] * 8, &lA[buf][sd[i]]);
#pragma unroll
    for (int i = 0; i < 4; ++i)
      async_b128(Bm + (size_t)(bn + srow[i]) * K + k0 + sq[i] * 8, &lB[buf][sd[i]]);
  };

  auto compute = [&](const unsigned int* bufA, const unsigned int* bufB) {
#pragma unroll
    for (int kh = 0; kh < 2; ++kh) {          // two K=32 slices of the 64-tile
      const int kb = kh * 16;                  // dword offset of slice
      FragBF af[4], bf[2];
#pragma unroll
      for (int mt = 0; mt < 4; ++mt) {
        const unsigned int* src = &bufA[(wm * 64 + mt * 16 + lanelo) * LDSW + kb];
        const int kh2 = laneHalf * 4;
#pragma unroll
        for (int j = 0; j < 4; ++j) af[mt].u[j] = src[kh2 + j];         // K 0..15
#pragma unroll
        for (int j = 0; j < 4; ++j) af[mt].u[4 + j] = src[8 + kh2 + j]; // K 16..31
      }
#pragma unroll
      for (int nt = 0; nt < 2; ++nt) {
        const unsigned int* src = &bufB[(wn * 32 + nt * 16 + lanelo) * LDSW + kb];
#pragma unroll
        for (int j = 0; j < 4; ++j) bf[nt].u[j] = src[laneHalf * 4 + j];
#pragma unroll
        for (int j = 0; j < 4; ++j) bf[nt].u[4 + j] = src[8 + laneHalf * 4 + j];
      }
#pragma unroll
      for (int mt = 0; mt < 4; ++mt)
#pragma unroll
        for (int nt = 0; nt < 2; ++nt)
          acc[mt][nt] = __builtin_amdgcn_wmma_f32_16x16x32_bf16(
              false, af[mt].v, false, bf[nt].v, (short)0, acc[mt][nt], false, false);
    }
  };

  stage(0, 0);
  const int KT = K / BLK_K;
  for (int kt = 0; kt < KT; ++kt) {
    wait_async0();        // own async writes done (ASYNCcnt is per-wave)
    __syncthreads();      // -> every wave's writes to this buffer visible
    if (kt + 1 < KT) stage((kt + 1) & 1, (kt + 1) * BLK_K);
    compute(lA[kt & 1], lB[kt & 1]);
    // no trailing barrier: next-iter wait+barrier orders buffer reuse
  }

  // ---- epilogue: C/D layout lane(0..15)=N, vgpr r -> M=r(+8 for hi half) ----
#pragma unroll
  for (int mt = 0; mt < 4; ++mt) {
#pragma unroll
    for (int nt = 0; nt < 2; ++nt) {
#pragma unroll
      for (int r = 0; r < 8; ++r) {
        const int gm = bm + wm * 64 + mt * 16 + laneHalf * 8 + r;
        const int gn = bn + wn * 32 + nt * 16 + lanelo;
        const size_t idx = (size_t)gm * N + gn;
        const float v = acc[mt][nt][r];
        if constexpr (EPI == 0) {
          outF[idx] = v;
        } else if constexpr (EPI == 1) {
          outF[idx] = 1.0f / (1.0f + __expf(-v));
        } else if constexpr (EPI == 2) {
          const float t = v > 0.0f ? v : 0.0f;
          outB[idx] = f32_to_bf16(t * t);
        } else {
          outF[idx] = res[idx] + v;
        }
      }
    }
  }
}

// ---------------------------------------------------------------------------
// host: pipeline
// ---------------------------------------------------------------------------
extern "C" void kernel_launch(void* const* d_in, const int* in_sizes, int n_in,
                              void* d_out, int out_size, void* d_ws, size_t ws_size,
                              hipStream_t stream) {
  const float* x    = (const float*)d_in[0];
  const float* ln1w = (const float*)d_in[1];
  const float* ln1b = (const float*)d_in[2];
  const float* ln2w = (const float*)d_in[3];
  const float* ln2b = (const float*)d_in[4];
  const float* tmk  = (const float*)d_in[5];
  const float* tmv  = (const float*)d_in[6];
  const float* tmr  = (const float*)d_in[7];
  const float* td   = (const float*)d_in[8];
  const float* tf   = (const float*)d_in[9];
  const float* Wk   = (const float*)d_in[10];
  const float* Wv   = (const float*)d_in[11];
  const float* Wr   = (const float*)d_in[12];
  const float* Wo   = (const float*)d_in[13];
  const float* ftmk = (const float*)d_in[14];
  const float* ftmr = (const float*)d_in[15];
  const float* Wkey = (const float*)d_in[16];
  const float* Wrec = (const float*)d_in[17];
  const float* Wval = (const float*)d_in[18];
  float* out = (float*)d_out;

  // ---- workspace layout (~540 MB peak) ----
  char* p = (char*)d_ws;
  auto alloc = [&](size_t bytes) { char* q = p; p += (bytes + 255) & ~(size_t)255; return q; };
  float*          h     = (float*)alloc((size_t)NROWS * CC * 4);
  unsigned short* xk    = (unsigned short*)alloc((size_t)NROWS * CC * 2);
  unsigned short* xv    = (unsigned short*)alloc((size_t)NROWS * CC * 2);
  unsigned short* xr    = (unsigned short*)alloc((size_t)NROWS * CC * 2);
  float*          kbuf  = (float*)alloc((size_t)NROWS * CC * 4);   // reused as kv
  float*          vbuf  = (float*)alloc((size_t)NROWS * CC * 4);
  float*          srbuf = (float*)alloc((size_t)NROWS * CC * 4);   // reused as sigmoid(rec)
  unsigned short* rwkv  = (unsigned short*)alloc((size_t)NROWS * CC * 2);
  unsigned short* kk    = (unsigned short*)alloc((size_t)NROWS * FF * 2);
  unsigned short* WkB   = (unsigned short*)alloc((size_t)CC * CC * 2);
  unsigned short* WvB   = (unsigned short*)alloc((size_t)CC * CC * 2);
  unsigned short* WrB   = (unsigned short*)alloc((size_t)CC * CC * 2);
  unsigned short* WoB   = (unsigned short*)alloc((size_t)CC * CC * 2);
  unsigned short* WrecB = (unsigned short*)alloc((size_t)CC * CC * 2);
  unsigned short* WkeyB = (unsigned short*)alloc((size_t)FF * CC * 2);
  unsigned short* WvalB = (unsigned short*)alloc((size_t)CC * FF * 2);

  const int sqC4 = (CC * CC) / 4, sqB = (CC * CC) / 1024;
  cvt_bf16_kernel<<<sqB, 256, 0, stream>>>(Wk,   WkB,   sqC4);
  cvt_bf16_kernel<<<sqB, 256, 0, stream>>>(Wv,   WvB,   sqC4);
  cvt_bf16_kernel<<<sqB, 256, 0, stream>>>(Wr,   WrB,   sqC4);
  cvt_bf16_kernel<<<sqB, 256, 0, stream>>>(Wo,   WoB,   sqC4);
  cvt_bf16_kernel<<<sqB, 256, 0, stream>>>(Wrec, WrecB, sqC4);
  cvt_bf16_kernel<<<sqB * 4, 256, 0, stream>>>(Wkey, WkeyB, sqC4 * 4);
  cvt_bf16_kernel<<<sqB * 4, 256, 0, stream>>>(Wval, WvalB, sqC4 * 4);

  // ---- attention half ----
  ln_kernel<<<NROWS, 256, 0, stream>>>(x, ln1w, ln1b, h);
  mix_att_kernel<<<NROWS, 256, 0, stream>>>(h, tmk, tmv, tmr, xk, xv, xr);

  dim3 gC(CC / BLK_N, NROWS / BLK_M);   // (8, 128)
  gemm_nt_bf16<0><<<gC, 256, 0, stream>>>(xk, WkB, kbuf,  nullptr, nullptr, NROWS, CC, CC);
  gemm_nt_bf16<0><<<gC, 256, 0, stream>>>(xv, WvB, vbuf,  nullptr, nullptr, NROWS, CC, CC);
  gemm_nt_bf16<1><<<gC, 256, 0, stream>>>(xr, WrB, srbuf, nullptr, nullptr, NROWS, CC, CC);

  wkv_kernel<<<(BB * CC) / 256, 256, 0, stream>>>(td, tf, kbuf, vbuf, srbuf, rwkv);

  // x1 = x + rwkv @ Wo^T  -> d_out
  gemm_nt_bf16<3><<<gC, 256, 0, stream>>>(rwkv, WoB, out, nullptr, x, NROWS, CC, CC);

  // ---- FFN half ----
  ln_kernel<<<NROWS, 256, 0, stream>>>(out, ln2w, ln2b, h);
  mix_ffn_kernel<<<NROWS, 256, 0, stream>>>(h, ftmk, ftmr, xk, xr);

  dim3 gF(FF / BLK_N, NROWS / BLK_M);   // (32, 128)
  gemm_nt_bf16<2><<<gF, 256, 0, stream>>>(xk, WkeyB, nullptr, kk, nullptr, NROWS, FF, CC);
  gemm_nt_bf16<0><<<gC, 256, 0, stream>>>(kk, WvalB, kbuf,  nullptr, nullptr, NROWS, CC, FF);
  gemm_nt_bf16<1><<<gC, 256, 0, stream>>>(xr, WrecB, srbuf, nullptr, nullptr, NROWS, CC, CC);

  final_kernel<<<(NROWS * CC) / 1024, 256, 0, stream>>>(out, srbuf, kbuf);
}
